// TransformerBlockTPU_63282048139681
// MI455X (gfx1250) — matrix operand order repaired and checked by reading, not compiled, not run on hardware
//
#include <hip/hip_runtime.h>
#include <cstddef>
#include <cstdint>

// ---------------------------------------------------------------------------
// Types for CDNA5 WMMA (wave32): v16bf A/B fragments, v8f C/D accumulators.
// ---------------------------------------------------------------------------
typedef __bf16 v16bf __attribute__((ext_vector_type(16)));
typedef float  v8f   __attribute__((ext_vector_type(8)));
typedef unsigned int v4u __attribute__((ext_vector_type(4)));
typedef int  v8i __attribute__((ext_vector_type(8)));
typedef int  v4i __attribute__((ext_vector_type(4)));
typedef unsigned short u16;

union Frag {               // 32 bytes: one 16-bf16 WMMA operand
    uint4 u[2];
    v16bf v;
};

#define WMMA_BF16(A, B, C) \
    __builtin_amdgcn_wmma_f32_16x16x32_bf16(false, (A), false, (B), (short)0, (C), false, false)

#if defined(__gfx1250__) && __has_builtin(__builtin_amdgcn_tensor_load_to_lds) && \
    __has_builtin(__builtin_amdgcn_s_wait_tensorcnt)
#define HAVE_TDM 1
#else
#define HAVE_TDM 0
#endif

__device__ __forceinline__ u16 f2bf(float f) {
    unsigned int x = __float_as_uint(f);
    unsigned int lsb = (x >> 16) & 1u;
    x += 0x7fffu + lsb;                 // round-to-nearest-even
    return (u16)(x >> 16);
}

// ---------------------------------------------------------------------------
// Problem constants
// ---------------------------------------------------------------------------
constexpr int B_  = 2;
constexpr int S_  = 2048;
constexpr int D_  = 2048;
constexpr int H_  = 16;
constexpr int HD_ = 128;
constexpr int M_  = 8192;
constexpr int ROWS = B_ * S_;           // 4096 flattened token rows
constexpr int LDT = 40;                 // LDS row stride (bf16): 80B, 16B-aligned, bank-skewed

#if HAVE_TDM
// LDS byte offset of a __shared__ object (generic -> AS3 -> integer)
__device__ __forceinline__ unsigned lds_off(const void* p) {
    return (unsigned)(unsigned long long)(__attribute__((address_space(3))) const void*)p;
}

// Issue one TDM 2-D tile load: 128 rows x 32 bf16, row stride = strideElems,
// into LDS with pad_interval=16 DWORDs / pad_amount=4 DWORDs => LDS row
// stride of 40 bf16 (matches LDT). Descriptor per CDNA5 ISA ch.8 (D# groups).
// This toolchain exposes the 6-arg builtin: (g0, g1, g2, g3, g_extra8, cpol).
__device__ __forceinline__ void tdm_load_tile_128x32(const u16* g, unsigned ldsByteAddr,
                                                     int strideElems) {
    unsigned long long ga = (unsigned long long)(uintptr_t)g;
    v4u g0;
    g0.x = 1u;                                   // count=1, load, no gather
    g0.y = ldsByteAddr;                          // lds_addr [63:32]
    g0.z = (unsigned)ga;                         // global_addr [95:64]
    g0.w = (unsigned)((ga >> 32) & 0x1ffffffu)   // global_addr [120:96]
         | (2u << 30);                           // type=2 ("image") [127:126]
    const unsigned td0 = 1u << 20, td1 = 1u << 20;   // generous tensor dims (OOB only)
    v8i g1;
    g1[0] = (int)((1u << 16)        // data_size = 1 -> 2 bytes
                | (1u << 20)        // pad_enable
                | (3u << 22)        // pad_interval: 16 DWORDs
                | (3u << 25));      // pad_amount:   4 DWORDs
    g1[1] = (int)((td0 & 0xffffu) << 16);                    // tensor_dim0[15:0]
    g1[2] = (int)((td0 >> 16) | ((td1 & 0xffffu) << 16));    // dim0[31:16] | dim1[15:0]
    g1[3] = (int)((td1 >> 16) | (32u << 16));                // dim1[31:16] | tile_dim0=32
    g1[4] = 128;                                             // tile_dim1=128, tile_dim2=0
    g1[5] = strideElems;                                     // tensor_dim0_stride[31:0]
    g1[6] = 0;                                               // stride hi, dim1_stride lo
    g1[7] = 0;
    v4i gz4 = {0, 0, 0, 0};
    v8i gz8 = {0, 0, 0, 0, 0, 0, 0, 0};
    __builtin_amdgcn_tensor_load_to_lds(g0, g1, gz4, gz4, gz8, 0);
}
#endif

// ---------------------------------------------------------------------------
// Weight convert + transpose: out[n*K + k] = bf16(in[k*N + n])
// ---------------------------------------------------------------------------
__global__ void conv_transpose_kernel(const float* __restrict__ in, u16* __restrict__ out,
                                      int K, int N) {
    size_t idx = (size_t)blockIdx.x * blockDim.x + threadIdx.x;
    size_t n = idx / (size_t)K;
    size_t k = idx % (size_t)K;
    out[idx] = f2bf(in[k * (size_t)N + n]);
}

// ---------------------------------------------------------------------------
// RMSNorm (f32 in, bf16 out); one block per row
// ---------------------------------------------------------------------------
__global__ __launch_bounds__(256) void rmsnorm_kernel(const float* __restrict__ x,
                                                      const float* __restrict__ scale,
                                                      u16* __restrict__ out) {
    __shared__ float red[256];
    int row = blockIdx.x;
    const float* xr = x + (size_t)row * D_;
    float ss = 0.f;
    for (int i = threadIdx.x; i < D_; i += 256) { float v = xr[i]; ss += v * v; }
    red[threadIdx.x] = ss;
    __syncthreads();
    for (int s = 128; s > 0; s >>= 1) {
        if ((int)threadIdx.x < s) red[threadIdx.x] += red[threadIdx.x + s];
        __syncthreads();
    }
    float inv = rsqrtf(red[0] / (float)D_ + 1e-6f);
    for (int i = threadIdx.x; i < D_; i += 256)
        out[(size_t)row * D_ + i] = f2bf(xr[i] * inv * scale[i]);
}

// ---------------------------------------------------------------------------
// Generic bf16 GEMM: C[ROWS x Ndim] = A[ROWS x Kdim] * Bt^T, Bt stored [Ndim x Kdim].
// Block tile 128x128, 8 waves (2x4), wave tile 64x32, K-step 32.
// TDM path: double-buffered LDS; wave0 DMAs the A tile, wave1 the B tile,
// overlap with WMMA; completion via TENSORcnt + workgroup barrier.
// EPI 1: QKV scatter (Q,K row-major [b][h][s][d]; V transposed [b][h][d][s])
// EPI 2: f32 out = acc + resid
// ---------------------------------------------------------------------------
template <int EPI>
__global__ __launch_bounds__(256) void gemm128_kernel(
    const u16* __restrict__ A, const u16* __restrict__ Bt, int Kdim, int Ndim,
    const float* __restrict__ resid, float* __restrict__ outF,
    u16* __restrict__ qOut, u16* __restrict__ kOut, u16* __restrict__ vtOut) {
#if HAVE_TDM
    __shared__ __align__(16) u16 lA[2][128 * LDT];
    __shared__ __align__(16) u16 lB[2][128 * LDT];
#else
    __shared__ __align__(16) u16 lA[1][128 * LDT];
    __shared__ __align__(16) u16 lB[1][128 * LDT];
#endif

    const int tid  = threadIdx.x;
    const int lane = tid & 31;
    const int w    = tid >> 5;
    const int hi   = lane >> 4;
    const int l16  = lane & 15;
    const int wm   = w >> 2;            // 0..1  (64 rows each)
    const int wn   = w & 3;             // 0..3  (32 cols each)
    const int m0   = blockIdx.x * 128;
    const int n0   = blockIdx.y * 128;

    v8f zero = {};
    v8f acc[4][2];
#pragma unroll
    for (int i = 0; i < 4; ++i)
#pragma unroll
        for (int j = 0; j < 2; ++j) acc[i][j] = zero;

    const int nk = Kdim >> 5;

#if HAVE_TDM
    // Scalar (wave-uniform) branch: TDM ignores EXEC, so guard must be SALU.
    const int wsc = __builtin_amdgcn_readfirstlane(w);
    if (wsc == 0) tdm_load_tile_128x32(A + (size_t)m0 * Kdim, lds_off(&lA[0][0]), Kdim);
    if (wsc == 1) tdm_load_tile_128x32(Bt + (size_t)n0 * Kdim, lds_off(&lB[0][0]), Kdim);
#endif

    for (int kb = 0; kb < nk; ++kb) {
        const int k0 = kb << 5;
#if HAVE_TDM
        const int cur = kb & 1, nxt = cur ^ 1;
        const bool more = (kb + 1) < nk;
        if (wsc == 0) {
            if (more) {
                tdm_load_tile_128x32(A + (size_t)m0 * Kdim + k0 + 32, lds_off(&lA[nxt][0]), Kdim);
                __builtin_amdgcn_s_wait_tensorcnt(1);   // tile kb done (TDM in-order)
            } else {
                __builtin_amdgcn_s_wait_tensorcnt(0);
            }
        }
        if (wsc == 1) {
            if (more) {
                tdm_load_tile_128x32(Bt + (size_t)n0 * Kdim + k0 + 32, lds_off(&lB[nxt][0]), Kdim);
                __builtin_amdgcn_s_wait_tensorcnt(1);
            } else {
                __builtin_amdgcn_s_wait_tensorcnt(0);
            }
        }
        __syncthreads();
        const u16* sA = &lA[cur][0];
        const u16* sB = &lB[cur][0];
#else
        const int srow = tid >> 1;
        const int schk = (tid & 1) * 16;
        const uint4* ga = reinterpret_cast<const uint4*>(A + (size_t)(m0 + srow) * Kdim + k0 + schk);
        uint4 a0 = ga[0], a1 = ga[1];
        const uint4* gb = reinterpret_cast<const uint4*>(Bt + (size_t)(n0 + srow) * Kdim + k0 + schk);
        uint4 b0 = gb[0], b1 = gb[1];
        *reinterpret_cast<uint4*>(&lA[0][srow * LDT + schk])     = a0;
        *reinterpret_cast<uint4*>(&lA[0][srow * LDT + schk + 8]) = a1;
        *reinterpret_cast<uint4*>(&lB[0][srow * LDT + schk])     = b0;
        *reinterpret_cast<uint4*>(&lB[0][srow * LDT + schk + 8]) = b1;
        __syncthreads();
        const u16* sA = &lA[0][0];
        const u16* sB = &lB[0][0];
#endif

        Frag af[4], bf[2];
#pragma unroll
        for (int sm = 0; sm < 4; ++sm) {
            int r = wm * 64 + sm * 16 + l16;                       // A row (M)
            af[sm].u[0] = *reinterpret_cast<const uint4*>(&sA[r * LDT + 8 * hi]);
            af[sm].u[1] = *reinterpret_cast<const uint4*>(&sA[r * LDT + 16 + 8 * hi]);
        }
#pragma unroll
        for (int sn = 0; sn < 2; ++sn) {
            int r = wn * 32 + sn * 16 + l16;                       // B row (N)
            bf[sn].u[0] = *reinterpret_cast<const uint4*>(&sB[r * LDT + 8 * hi]);
            bf[sn].u[1] = *reinterpret_cast<const uint4*>(&sB[r * LDT + 16 + 8 * hi]);
        }
#pragma unroll
        for (int sm = 0; sm < 4; ++sm)
#pragma unroll
            for (int sn = 0; sn < 2; ++sn)
                acc[sm][sn] = WMMA_BF16(af[sm].v, bf[sn].v, acc[sm][sn]);
        __syncthreads();
    }

    // Epilogue: C-layout lane L holds (M = r + 8*hi, N = l16) for r = 0..7
#pragma unroll
    for (int sm = 0; sm < 4; ++sm)
#pragma unroll
        for (int sn = 0; sn < 2; ++sn) {
            int gmb = m0 + wm * 64 + sm * 16 + 8 * hi;
            int gn  = n0 + wn * 32 + sn * 16 + l16;
#pragma unroll
            for (int r = 0; r < 8; ++r) {
                int m = gmb + r;
                float v = acc[sm][sn][r];
                if constexpr (EPI == 1) {
                    int which = gn >> 11;           // n / 2048 : q,k,v
                    int rem   = gn & 2047;
                    int h     = rem >> 7;           // / HD
                    int d     = rem & 127;
                    int b     = m >> 11;            // / S
                    int s     = m & 2047;
                    size_t bh = (size_t)b * H_ + h;
                    if (which == 0)      qOut[(bh * S_ + s) * HD_ + d]  = f2bf(v);
                    else if (which == 1) kOut[(bh * S_ + s) * HD_ + d]  = f2bf(v);
                    else                 vtOut[(bh * HD_ + d) * S_ + s] = f2bf(v);
                } else {
                    outF[(size_t)m * Ndim + gn] = v + resid[(size_t)m * Ndim + gn];
                }
            }
        }
}

// ---------------------------------------------------------------------------
// Flash attention: 1 wave per (b, h, 16-query block). Causal, online softmax.
// Q,K stored [b][h][s][d]; V stored transposed [b][h][d][s] so PV K-dim loads
// are contiguous. P goes through per-wave LDS to convert C-layout -> A-layout.
// ---------------------------------------------------------------------------
__global__ __launch_bounds__(256) void attention_kernel(
    const u16* __restrict__ q, const u16* __restrict__ k, const u16* __restrict__ vt,
    u16* __restrict__ attnOut) {
    __shared__ __align__(16) u16 lP[8][16 * LDT];

    const int lane = threadIdx.x & 31;
    const int w    = threadIdx.x >> 5;
    const int hi   = lane >> 4;
    const int l16  = lane & 15;

    const int task = blockIdx.x * 8 + w;       // 4096 tasks
    const int qb   = task & 127;               // query block (S/16)
    const int h    = (task >> 7) & 15;
    const int b    = task >> 11;

    const size_t bh = (size_t)b * H_ + h;
    const u16* qbase = q  + (bh * S_ + (size_t)qb * 16) * HD_;
    const u16* kbase = k  + bh * S_ * HD_;
    const u16* vbase = vt + bh * (size_t)HD_ * S_;

    // Q fragments: A-layout, 4 chunks of K=32 over HD=128
    Frag aQ[4];
#pragma unroll
    for (int c = 0; c < 4; ++c) {
        const u16* row = qbase + (size_t)l16 * HD_ + c * 32;
        aQ[c].u[0] = *reinterpret_cast<const uint4*>(row + 8 * hi);
        aQ[c].u[1] = *reinterpret_cast<const uint4*>(row + 16 + 8 * hi);
    }

    v8f zero = {};
    v8f o[8];
#pragma unroll
    for (int t = 0; t < 8; ++t) o[t] = zero;
    float mrow[8], lrow[8];
#pragma unroll
    for (int r = 0; r < 8; ++r) { mrow[r] = -1e30f; lrow[r] = 0.f; }

    const float scale = 0.08838834764831845f;  // 1/sqrt(128)
    const int nkb = ((qb * 16 + 15) >> 5) + 1; // causal: 32-key blocks

    for (int kb = 0; kb < nkb; ++kb) {
        const int kk = kb * 32;
        v8f s0 = zero, s1 = zero;
#pragma unroll
        for (int c = 0; c < 4; ++c) {
            const int d0 = c * 32;
            Frag b0, b1;
            const u16* kr0 = kbase + (size_t)(kk + l16) * HD_ + d0 + 8 * hi;
            b0.u[0] = *reinterpret_cast<const uint4*>(kr0);
            b0.u[1] = *reinterpret_cast<const uint4*>(kr0 + 16);
            const u16* kr1 = kbase + (size_t)(kk + 16 + l16) * HD_ + d0 + 8 * hi;
            b1.u[0] = *reinterpret_cast<const uint4*>(kr1);
            b1.u[1] = *reinterpret_cast<const uint4*>(kr1 + 16);
            s0 = WMMA_BF16(aQ[c].v, b0.v, s0);
            s1 = WMMA_BF16(aQ[c].v, b1.v, s1);
        }

        float p0[8], p1[8], alpha[8];
#pragma unroll
        for (int r = 0; r < 8; ++r) {
            int qi  = qb * 16 + r + 8 * hi;
            int k0i = kk + l16;
            int k1i = kk + 16 + l16;
            float v0 = (k0i <= qi) ? s0[r] * scale : -1e30f;
            float v1 = (k1i <= qi) ? s1[r] * scale : -1e30f;
            float mx = fmaxf(v0, v1);
            for (int msk = 1; msk < 16; msk <<= 1)        // row-wise over 16-lane half
                mx = fmaxf(mx, __shfl_xor(mx, msk, 32));
            float mnew = fmaxf(mrow[r], mx);
            alpha[r] = __expf(mrow[r] - mnew);
            p0[r] = __expf(v0 - mnew);
            p1[r] = __expf(v1 - mnew);
            float sum = p0[r] + p1[r];
            for (int msk = 1; msk < 16; msk <<= 1)
                sum += __shfl_xor(sum, msk, 32);
            lrow[r] = lrow[r] * alpha[r] + sum;
            mrow[r] = mnew;
        }
#pragma unroll
        for (int t = 0; t < 8; ++t)
#pragma unroll
            for (int r = 0; r < 8; ++r) o[t][r] *= alpha[r];

        // C-layout P -> LDS -> A-layout fragment (DS in-order within wave)
        u16* lp = &lP[w][0];
#pragma unroll
        for (int r = 0; r < 8; ++r) {
            int row = r + 8 * hi;
            lp[row * LDT + l16]      = f2bf(p0[r]);
            lp[row * LDT + 16 + l16] = f2bf(p1[r]);
        }
        asm volatile("s_wait_dscnt 0" ::: "memory");
        Frag aP;
        aP.u[0] = *reinterpret_cast<const uint4*>(&lp[l16 * LDT + 8 * hi]);
        aP.u[1] = *reinterpret_cast<const uint4*>(&lp[l16 * LDT + 16 + 8 * hi]);

#pragma unroll
        for (int t = 0; t < 8; ++t) {
            Frag bV;
            const u16* vr = vbase + (size_t)(t * 16 + l16) * S_ + kk + 8 * hi;
            bV.u[0] = *reinterpret_cast<const uint4*>(vr);
            bV.u[1] = *reinterpret_cast<const uint4*>(vr + 16);
            o[t] = WMMA_BF16(aP.v, bV.v, o[t]);
        }
    }

    // Finalize: attn[(b*S + q) * (H*HD) + h*HD + d]
#pragma unroll
    for (int r = 0; r < 8; ++r) {
        float inv = 1.f / lrow[r];
        int qi = qb * 16 + r + 8 * hi;
        size_t base = ((size_t)b * S_ + qi) * (H_ * HD_) + (size_t)h * HD_;
#pragma unroll
        for (int t = 0; t < 8; ++t)
            attnOut[base + t * 16 + l16] = f2bf(o[t][r] * inv);
    }
}

// ---------------------------------------------------------------------------
// Fused gate/up GEMM + silu(g)*u epilogue. Block tile 128x64, 8 waves (4x2),
// wave tile 32x32 per matrix.
// ---------------------------------------------------------------------------
__global__ __launch_bounds__(256) void gemm_gateup_kernel(
    const u16* __restrict__ A, const u16* __restrict__ Bg, const u16* __restrict__ Bu,
    u16* __restrict__ act) {
    const int Kdim = D_;
    __shared__ __align__(16) u16 lA[128 * LDT];
    __shared__ __align__(16) u16 lG[64 * LDT];
    __shared__ __align__(16) u16 lU[64 * LDT];

    const int tid  = threadIdx.x;
    const int lane = tid & 31;
    const int w    = tid >> 5;
    const int hi   = lane >> 4;
    const int l16  = lane & 15;
    const int wm   = w >> 1;            // 0..3 (32 rows)
    const int wn   = w & 1;             // 0..1 (32 cols)
    const int m0   = blockIdx.x * 128;
    const int n0   = blockIdx.y * 64;
    const int srow = tid >> 1;
    const int schk = (tid & 1) * 16;

    v8f zero = {};
    v8f g[2][2], u[2][2];
#pragma unroll
    for (int i = 0; i < 2; ++i)
#pragma unroll
        for (int j = 0; j < 2; ++j) { g[i][j] = zero; u[i][j] = zero; }

    for (int k0 = 0; k0 < Kdim; k0 += 32) {
        const uint4* ga = reinterpret_cast<const uint4*>(A + (size_t)(m0 + srow) * Kdim + k0 + schk);
        uint4 a0 = ga[0], a1 = ga[1];
        *reinterpret_cast<uint4*>(&lA[srow * LDT + schk])     = a0;
        *reinterpret_cast<uint4*>(&lA[srow * LDT + schk + 8]) = a1;
        if (tid < 128) {
            int brow = tid >> 1, bchk = (tid & 1) * 16;
            const uint4* gb = reinterpret_cast<const uint4*>(Bg + (size_t)(n0 + brow) * Kdim + k0 + bchk);
            uint4 b0 = gb[0], b1 = gb[1];
            *reinterpret_cast<uint4*>(&lG[brow * LDT + bchk])     = b0;
            *reinterpret_cast<uint4*>(&lG[brow * LDT + bchk + 8]) = b1;
        } else {
            int t2 = tid - 128;
            int brow = t2 >> 1, bchk = (t2 & 1) * 16;
            const uint4* gb = reinterpret_cast<const uint4*>(Bu + (size_t)(n0 + brow) * Kdim + k0 + bchk);
            uint4 b0 = gb[0], b1 = gb[1];
            *reinterpret_cast<uint4*>(&lU[brow * LDT + bchk])     = b0;
            *reinterpret_cast<uint4*>(&lU[brow * LDT + bchk + 8]) = b1;
        }
        __syncthreads();

        Frag af[2], gf[2], uf[2];
#pragma unroll
        for (int sm = 0; sm < 2; ++sm) {
            int r = wm * 32 + sm * 16 + l16;
            af[sm].u[0] = *reinterpret_cast<const uint4*>(&lA[r * LDT + 8 * hi]);
            af[sm].u[1] = *reinterpret_cast<const uint4*>(&lA[r * LDT + 16 + 8 * hi]);
        }
#pragma unroll
        for (int sn = 0; sn < 2; ++sn) {
            int r = wn * 32 + sn * 16 + l16;
            gf[sn].u[0] = *reinterpret_cast<const uint4*>(&lG[r * LDT + 8 * hi]);
            gf[sn].u[1] = *reinterpret_cast<const uint4*>(&lG[r * LDT + 16 + 8 * hi]);
            uf[sn].u[0] = *reinterpret_cast<const uint4*>(&lU[r * LDT + 8 * hi]);
            uf[sn].u[1] = *reinterpret_cast<const uint4*>(&lU[r * LDT + 16 + 8 * hi]);
        }
#pragma unroll
        for (int sm = 0; sm < 2; ++sm)
#pragma unroll
            for (int sn = 0; sn < 2; ++sn) {
                g[sm][sn] = WMMA_BF16(af[sm].v, gf[sn].v, g[sm][sn]);
                u[sm][sn] = WMMA_BF16(af[sm].v, uf[sn].v, u[sm][sn]);
            }
        __syncthreads();
    }

#pragma unroll
    for (int sm = 0; sm < 2; ++sm)
#pragma unroll
        for (int sn = 0; sn < 2; ++sn) {
            int gmb = m0 + wm * 32 + sm * 16 + 8 * hi;
            int gn  = n0 + wn * 32 + sn * 16 + l16;
#pragma unroll
            for (int r = 0; r < 8; ++r) {
                float gv = g[sm][sn][r];
                float uv = u[sm][sn][r];
                float silu = gv / (1.f + __expf(-gv));
                act[(size_t)(gmb + r) * M_ + gn] = f2bf(silu * uv);
            }
        }
}

// ---------------------------------------------------------------------------
// Host-side orchestration
// ---------------------------------------------------------------------------
extern "C" void kernel_launch(void* const* d_in, const int* in_sizes, int n_in,
                              void* d_out, int out_size, void* d_ws, size_t ws_size,
                              hipStream_t stream) {
    (void)in_sizes; (void)n_in; (void)out_size; (void)ws_size;
    const float* x     = (const float*)d_in[0];
    const float* wqkv  = (const float*)d_in[1];
    const float* wo    = (const float*)d_in[2];
    const float* wgate = (const float*)d_in[3];
    const float* wup   = (const float*)d_in[4];
    const float* wdown = (const float*)d_in[5];
    const float* rms1  = (const float*)d_in[6];
    const float* rms2  = (const float*)d_in[7];
    float* out = (float*)d_out;

    // Workspace layout (bf16 unless noted)
    char* p = (char*)d_ws;
    auto take = [&](size_t bytes) { char* r = p; p += bytes; return r; };
    u16* wqkvT = (u16*)take((size_t)(3 * H_ * HD_) * D_ * 2);   // [6144][2048]
    u16* woT   = (u16*)take((size_t)D_ * D_ * 2);               // [2048][2048]
    u16* wgT   = (u16*)take((size_t)M_ * D_ * 2);               // [8192][2048]
    u16* wuT   = (u16*)take((size_t)M_ * D_ * 2);
    u16* wdT   = (u16*)take((size_t)D_ * M_ * 2);               // [2048][8192]
    u16* hn    = (u16*)take((size_t)ROWS * D_ * 2);
    u16* qbuf  = (u16*)take((size_t)B_ * H_ * S_ * HD_ * 2);
    u16* kbuf  = (u16*)take((size_t)B_ * H_ * S_ * HD_ * 2);
    u16* vtbuf = (u16*)take((size_t)B_ * H_ * HD_ * S_ * 2);
    u16* attn  = (u16*)take((size_t)ROWS * D_ * 2);
    float* x1  = (float*)take((size_t)ROWS * D_ * 4);
    u16* h2n   = (u16*)take((size_t)ROWS * D_ * 2);
    u16* act   = (u16*)take((size_t)ROWS * M_ * 2);

    // 1) weight convert + transpose (f32 -> bf16, [K][N] -> [N][K])
    conv_transpose_kernel<<<(D_ * 3 * H_ * HD_) / 256, 256, 0, stream>>>(wqkv,  wqkvT, D_, 3 * H_ * HD_);
    conv_transpose_kernel<<<(D_ * D_) / 256,          256, 0, stream>>>(wo,    woT,   D_, D_);
    conv_transpose_kernel<<<(D_ * M_) / 256,          256, 0, stream>>>(wgate, wgT,   D_, M_);
    conv_transpose_kernel<<<(D_ * M_) / 256,          256, 0, stream>>>(wup,   wuT,   D_, M_);
    conv_transpose_kernel<<<(M_ * D_) / 256,          256, 0, stream>>>(wdown, wdT,   M_, D_);

    // 2) RMSNorm 1
    rmsnorm_kernel<<<ROWS, 256, 0, stream>>>(x, rms1, hn);

    // 3) QKV projection, scattering q/k (row-major) and v (transposed)
    gemm128_kernel<1><<<dim3(ROWS / 128, (3 * H_ * HD_) / 128), 256, 0, stream>>>(
        hn, wqkvT, D_, 3 * H_ * HD_, nullptr, nullptr, qbuf, kbuf, vtbuf);

    // 4) Causal flash attention (1 wave per 16-query block per head)
    attention_kernel<<<(B_ * H_ * (S_ / 16)) / 8, 256, 0, stream>>>(qbuf, kbuf, vtbuf, attn);

    // 5) O-projection + residual -> x1 (f32)
    gemm128_kernel<2><<<dim3(ROWS / 128, D_ / 128), 256, 0, stream>>>(
        attn, woT, D_, D_, x, x1, nullptr, nullptr, nullptr);

    // 6) RMSNorm 2
    rmsnorm_kernel<<<ROWS, 256, 0, stream>>>(x1, rms2, h2n);

    // 7) Fused gate/up + silu(g)*u -> act (bf16)
    gemm_gateup_kernel<<<dim3(ROWS / 128, M_ / 64), 256, 0, stream>>>(h2n, wgT, wuT, act);

    // 8) Down projection + residual -> out (f32)
    gemm128_kernel<2><<<dim3(ROWS / 128, D_ / 128), 256, 0, stream>>>(
        act, wdT, M_, D_, x1, out, nullptr, nullptr, nullptr);
}
